// KmeansVectorQuantizer_38001870635050
// MI455X (gfx1250) — matrix operand order repaired and checked by reading, not compiled, not run on hardware
//
#include <hip/hip_runtime.h>
#include <hip/hip_bf16.h>

typedef __attribute__((ext_vector_type(16))) __bf16 v16bf;
typedef __attribute__((ext_vector_type(8)))  float  v8f;
typedef int v4i __attribute__((vector_size(16)));

union FragAB { uint4 q[2]; v16bf v; };

#define BT      8192   // B*T
#define GD      1024   // G*D
#define NG      8
#define DD      128
#define VV      1024
#define IDS_N   (BT * NG)        // 65536
#define Q_N     (BT * GD)        // 8388608

#if __has_builtin(__builtin_amdgcn_global_load_async_to_lds_b128) && \
    __has_builtin(__builtin_amdgcn_s_wait_asynccnt)
#define HAVE_ASYNC 1
#else
#define HAVE_ASYNC 0
#endif

#if HAVE_ASYNC
typedef __attribute__((address_space(1))) v4i* gas_ptr;   // global int4*
typedef __attribute__((address_space(3))) v4i* las_ptr;   // LDS int4*
#endif

__device__ __forceinline__ unsigned short f2bf(float f) {
    unsigned int u = __float_as_uint(f);
    unsigned int r = u + 0x7FFFu + ((u >> 16) & 1u);   // RNE
    return (unsigned short)(r >> 16);
}

// ---------------- prep: codebook fp32 -> bf16 [g][v][d], c2[g][v], zero accum ----
__global__ void vq_prep(const float* __restrict__ cb,
                        unsigned short* __restrict__ cbBf,
                        float* __restrict__ c2,
                        float* __restrict__ lossAcc) {
    const int rowIdx = blockIdx.x;            // (v*8 + g), 0..8191
    const int v = rowIdx >> 3, g = rowIdx & 7;
    const int d = threadIdx.x;                // 128 threads
    float f = cb[(size_t)rowIdx * DD + d];
    cbBf[((size_t)g * VV + v) * DD + d] = f2bf(f);
    float s = f * f;
    #pragma unroll
    for (int off = 16; off > 0; off >>= 1) s += __shfl_xor(s, off, 32);
    __shared__ float part[4];
    if ((threadIdx.x & 31) == 0) part[threadIdx.x >> 5] = s;
    __syncthreads();
    if (threadIdx.x == 0) {
        c2[g * VV + v] = part[0] + part[1] + part[2] + part[3];
        if (rowIdx == 0) *lossAcc = 0.0f;
    }
}

// 16 KB codebook tile: global -> LDS, 4 x 16B chunks per thread
__device__ __forceinline__ void tileCopy(const unsigned short* __restrict__ src,
                                         unsigned short (* __restrict__ dst)[136],
                                         int tid) {
    unsigned short* s = const_cast<unsigned short*>(src);
    #pragma unroll
    for (int i = 0; i < 4; ++i) {
        int c   = tid + i * 256;
        int row = c >> 4;             // 0..63
        int col = (c & 15) << 3;      // 0..120 step 8
#if HAVE_ASYNC
        __builtin_amdgcn_global_load_async_to_lds_b128(
            (gas_ptr)(s + row * DD + col),
            (las_ptr)&dst[row][col], 0, 0);
#else
        *(uint4*)&dst[row][col] = *(const uint4*)(s + row * DD + col);
#endif
    }
}

__device__ __forceinline__ void tileWait() {
#if HAVE_ASYNC
    __builtin_amdgcn_s_wait_asynccnt(0);
#endif
    __syncthreads();
}

// ---------------- main: per (32-row tile, g): WMMA dist + argmin + gather -------
__global__ void __launch_bounds__(256)
vq_main(const float* __restrict__ inputs,
        const int*   __restrict__ paddings,
        const float* __restrict__ cbF,
        const unsigned short* __restrict__ cbBf,
        const float* __restrict__ c2,
        float* __restrict__ lossAcc,
        float* __restrict__ outIds,
        float* __restrict__ outQ) {
    const int tid  = threadIdx.x;
    const int g    = blockIdx.x >> 8;     // 0..7  (256 tiles share a codebook slice)
    const int tile = blockIdx.x & 255;    // 0..255
    const int btBase = tile * 32;
    const int lane = tid & 31;
    const int wid  = tid >> 5;
    const int wm   = wid >> 2;            // 0..1 : M sub-tile
    const int wn   = wid & 3;             // 0..3 : N sub-tile
    const int llo  = lane & 15;
    const int hi   = lane >> 4;

    __shared__ __align__(16) unsigned short sA[32][136];      // x tile bf16, padded
    __shared__ __align__(16) unsigned short sB[2][64][136];   // double-buffered cb tile
    __shared__ float sBestD[4][32];
    __shared__ int   sBestV[4][32];
    __shared__ int   sId[32];
    __shared__ float sLoss;

    if (tid == 0) sLoss = 0.0f;

    const float* c2g = c2 + g * VV;
    const unsigned short* cbSlice = cbBf + (size_t)g * VV * DD;

    // ---- stage x tile (32 x 128) to LDS as bf16, coalesced float4 reads ----
    for (int c = tid; c < 1024; c += 256) {
        int row = c >> 5;                 // 0..31
        int col = (c & 31) << 2;          // d, multiple of 4
        const float4 f4 = *(const float4*)(inputs +
            (size_t)(btBase + row) * GD + g * DD + col);
        unsigned int lo = (unsigned int)f2bf(f4.x) | ((unsigned int)f2bf(f4.y) << 16);
        unsigned int hw = (unsigned int)f2bf(f4.z) | ((unsigned int)f2bf(f4.w) << 16);
        *(uint2*)&sA[row][col] = make_uint2(lo, hw);
    }
    // kick off first codebook tile while sA settles
    tileCopy(cbSlice, sB[0], tid);
    __syncthreads();                      // sA ready

    // ---- A fragments: CDNA5 16-bit A layout (16x32), invariant across v ----
    FragAB aFrag[4];
    {
        int m = wm * 16 + llo;
        #pragma unroll
        for (int kc = 0; kc < 4; ++kc) {
            aFrag[kc].q[0] = *(const uint4*)&sA[m][kc * 32 + 8 * hi];       // e0..7
            aFrag[kc].q[1] = *(const uint4*)&sA[m][kc * 32 + 16 + 8 * hi];  // e8..15
        }
    }

    float bestD[8];
    int   bestV[8];
    #pragma unroll
    for (int r = 0; r < 8; ++r) { bestD[r] = 3.4e38f; bestV[r] = 0; }

    for (int vt = 0; vt < 16; ++vt) {
        tileWait();                       // sB[vt&1] ready on all waves
        if (vt + 1 < 16) {                // overlap next copy with compute
            tileCopy(cbSlice + (vt + 1) * 64 * DD, sB[(vt + 1) & 1], tid);
#if !HAVE_ASYNC
            __builtin_prefetch(cbSlice + (vt + 1) * 64 * DD + tid * 64, 0, 1);
#endif
        }
        const unsigned short (*buf)[136] = sB[vt & 1];

        // ---- load all 4 B fragments, then chain 4 WMMAs (K=128) ----
        FragAB bF[4];
        {
            int n = wn * 16 + llo;
            #pragma unroll
            for (int kc = 0; kc < 4; ++kc) {
                bF[kc].q[0] = *(const uint4*)&buf[n][kc * 32 + 8 * hi];
                bF[kc].q[1] = *(const uint4*)&buf[n][kc * 32 + 16 + 8 * hi];
            }
        }
        v8f acc = {};
        #pragma unroll
        for (int kc = 0; kc < 4; ++kc)
            acc = __builtin_amdgcn_wmma_f32_16x16x32_bf16(
                false, aFrag[kc].v, false, bF[kc].v, (short)0, acc, false, false);

        // ---- score = c2 - 2*xc (x2 is row-constant: irrelevant to argmin) ----
        int   v   = vt * 64 + wn * 16 + llo;
        float c2v = c2g[v];
        #pragma unroll
        for (int r = 0; r < 8; ++r) {
            float s = fmaf(-2.0f, acc[r], c2v);
            if (s < bestD[r]) { bestD[r] = s; bestV[r] = v; }
        }
    }

    // ---- argmin over N within each 16-lane half (tie -> lowest v) ----
    #pragma unroll
    for (int r = 0; r < 8; ++r) {
        #pragma unroll
        for (int off = 8; off > 0; off >>= 1) {
            float od = __shfl_xor(bestD[r], off, 32);
            int   ov = __shfl_xor(bestV[r], off, 32);
            if (od < bestD[r] || (od == bestD[r] && ov < bestV[r])) {
                bestD[r] = od; bestV[r] = ov;
            }
        }
        if (llo == 0) {
            int row = wm * 16 + r + 8 * hi;   // C-matrix M mapping
            sBestD[wn][row] = bestD[r];
            sBestV[wn][row] = bestV[r];
        }
    }
    __syncthreads();

    // ---- combine the 4 N-waves; emit ids ----
    if (tid < 32) {
        int row = tid;
        float bd = sBestD[0][row];
        int   bv = sBestV[0][row];
        #pragma unroll
        for (int w2 = 1; w2 < 4; ++w2) {
            float od = sBestD[w2][row]; int ov = sBestV[w2][row];
            if (od < bd || (od == bd && ov < bv)) { bd = od; bv = ov; }
        }
        sId[row] = bv;
        int bt = btBase + row;
        outIds[bt * NG + g] = paddings[bt] ? -1.0f : (float)bv;
    }
    __syncthreads();

    // ---- gather quantized (fp32), masked output, loss numerator ----
    {
        int row = tid >> 3;
        int d0  = (tid & 7) << 4;         // 16 floats per thread
        int bt  = btBase + row;
        int id  = sId[row];
        int pad = paddings[bt];
        const float* q = cbF    + ((size_t)id * NG + g) * DD + d0;
        const float* x = inputs + (size_t)bt * GD + g * DD + d0;
        float*       o = outQ   + (size_t)bt * GD + g * DD + d0;
        float lsum = 0.0f;
        #pragma unroll
        for (int c4 = 0; c4 < 4; ++c4) {
            float4 qv = *(const float4*)(q + c4 * 4);
            float4 xv = *(const float4*)(x + c4 * 4);
            float dx = qv.x - xv.x, dy = qv.y - xv.y;
            float dz = qv.z - xv.z, dw = qv.w - xv.w;
            lsum += dx * dx + dy * dy + dz * dz + dw * dw;
            float4 ov = pad ? make_float4(0.f, 0.f, 0.f, 0.f) : qv;
            *(float4*)(o + c4 * 4) = ov;
        }
        if (!pad) atomicAdd(&sLoss, lsum);
    }
    __syncthreads();
    if (tid == 0) atomicAdd(lossAcc, sLoss);
}

// ---------------- finalize: msum, scalar losses --------------------------------
__global__ void vq_fin(const int* __restrict__ paddings,
                       const float* __restrict__ lossAcc,
                       float* __restrict__ outLoss) {
    __shared__ float part[8];
    int t = threadIdx.x;
    int cnt = 0;
    for (int i = t; i < BT; i += 256) cnt += paddings[i];
    float c = (float)cnt;
    #pragma unroll
    for (int off = 16; off > 0; off >>= 1) c += __shfl_xor(c, off, 32);
    if ((t & 31) == 0) part[t >> 5] = c;
    __syncthreads();
    if (t == 0) {
        float tot = 0.f;
        #pragma unroll
        for (int i = 0; i < 8; ++i) tot += part[i];
        float msum = (float)BT - tot;
        float loss = (msum > 0.f) ? (*lossAcc) / msum : 0.f;
        outLoss[0] = loss;          // kmeans_loss  (forward-equal)
        outLoss[1] = loss;          // commitment_loss
        outLoss[2] = 2.0f * loss;   // total (BETA = 1)
    }
}

extern "C" void kernel_launch(void* const* d_in, const int* in_sizes, int n_in,
                              void* d_out, int out_size, void* d_ws, size_t ws_size,
                              hipStream_t stream) {
    const float* inputs   = (const float*)d_in[0];
    const int*   paddings = (const int*)  d_in[1];
    const float* codebook = (const float*)d_in[2];

    float* out     = (float*)d_out;
    float* outIds  = out;                       // 65536 ids (as float, -1 padded)
    float* outQ    = out + IDS_N;               // 8388608 quantized_st
    float* outLoss = out + IDS_N + Q_N;         // 3 scalars

    // workspace layout: [0] loss accum | [16..16+8192) c2 | bf16 codebook (2 MB)
    float* wsF = (float*)d_ws;
    float* lossAcc = wsF;
    float* c2      = wsF + 16;
    unsigned short* cbBf = (unsigned short*)(wsF + 16 + NG * VV);

    vq_prep<<<VV * NG, 128, 0, stream>>>(codebook, cbBf, c2, lossAcc);
    vq_main<<<256 * NG, 256, 0, stream>>>(inputs, paddings, codebook, cbBf, c2,
                                          lossAcc, outIds, outQ);
    vq_fin<<<1, 256, 0, stream>>>(paddings, lossAcc, outLoss);
}
